// AdditiveAttention_5540507812400
// MI455X (gfx1250) — matrix operand order repaired and checked by reading, not compiled, not run on hardware
//
#include <hip/hip_runtime.h>
#include <hip/hip_bf16.h>

// Problem constants (match reference)
constexpr int cB  = 8;
constexpr int cLQ = 256;
constexpr int cLK = 1024;
constexpr int cD  = 512;   // DQ == DK == DV
constexpr int cH  = 128;
constexpr float cNEG = -1000000.0f;

typedef __attribute__((ext_vector_type(16))) __bf16 v16bf;
typedef __attribute__((ext_vector_type(8)))  float  v8f;
typedef __attribute__((ext_vector_type(4)))  unsigned u32x4;
typedef __attribute__((ext_vector_type(8)))  unsigned u32x8;

__device__ __forceinline__ float fast_tanh(float x) {
#if __has_builtin(__builtin_amdgcn_tanhf)
    return __builtin_amdgcn_tanhf(x);      // v_tanh_f32 (gfx1250 TRANS op)
#else
    return tanhf(x);
#endif
}

__device__ __forceinline__ float fast_exp2(float x) {
#if __has_builtin(__builtin_amdgcn_exp2f)
    return __builtin_amdgcn_exp2f(x);      // v_exp_f32
#else
    return exp2f(x);
#endif
}

// CDNA5 async global->LDS copy (ASYNCcnt-tracked), ISA 15.18.3 opcode 98.
__device__ __forceinline__ void async_copy_b128(void* lds, const void* gmem) {
    unsigned l = (unsigned)(uintptr_t)lds;   // low 32 bits = LDS byte address
    asm volatile("global_load_async_to_lds_b128 %0, %1, off"
                 :: "v"(l), "v"(gmem) : "memory");
}
#define S_WAIT_ASYNCCNT(n) asm volatile("s_wait_asynccnt " #n ::: "memory")

template <short N>
__device__ __forceinline__ void wait_tensorcnt() {
#if __has_builtin(__builtin_amdgcn_s_wait_tensorcnt)
    __builtin_amdgcn_s_wait_tensorcnt(N);  // N is a constant expression here
#else
    asm volatile("s_wait_tensorcnt %0" :: "i"(N) : "memory");
#endif
}

// ---------------------------------------------------------------------------
// TDM: DMA one 64x128-f32 tile (row stride 128 floats in memory) into LDS,
// padding each 128-DWORD row with 4 DWORDs (-> 132-float LDS pitch, which
// breaks the 64-bank alignment for conflict-free strided reads).
// D# per CDNA5 ISA; 2-operand (<=2D) asm form.
// ---------------------------------------------------------------------------
__device__ __forceinline__ void tdm_load_tile(void* lds, const float* gsrc) {
    const unsigned lds_addr = (unsigned)(uintptr_t)lds;
    const unsigned long long ga = (unsigned long long)(uintptr_t)gsrc;
    u32x4 g0;
    g0[0] = 1u;                                            // count=1, user D#
    g0[1] = lds_addr;                                      // lds_addr (bytes)
    g0[2] = (unsigned)ga;                                  // global_addr[31:0]
    g0[3] = ((unsigned)(ga >> 32) & 0x01FFFFFFu)           // global_addr[56:32]
          | (2u << 30);                                    // type = 2 ("image")
    u32x8 g1;
    g1[0] = (2u << 16)     // data_size = 4B
          | (1u << 20)     // pad_enable
          | (6u << 22)     // pad_interval: 2^(6+1)=128 DWORDs between pads
          | (3u << 25);    // pad_amount: 3+1 = 4 DWORDs
    g1[1] = (128u << 16);  // tensor_dim0 = 128 (low 16 in [31:16])
    g1[2] = (1024u << 16); // tensor_dim0 hi = 0; tensor_dim1 = 1024 (low 16)
    g1[3] = (128u << 16);  // tensor_dim1 hi = 0; tile_dim0 = 128
    g1[4] = 64u;           // tile_dim1 = 64; tile_dim2 = 0
    g1[5] = 128u;          // tensor_dim0_stride[31:0] = 128
    g1[6] = 0u;            // stride0 hi, stride1 lo
    g1[7] = 0u;            // stride1 hi
    asm volatile("tensor_load_to_lds %0, %1" :: "s"(g0), "s"(g1) : "memory");
}

// ---------------------------------------------------------------------------
// Kernel 1: row-major GEMM  Y[nrows,N] = X[nrows,Kd] @ W[Kd,N]  via bf16 WMMA.
// One wave per 64x16 output strip: one B fragment feeds 4 WMMAs.
// ---------------------------------------------------------------------------
__global__ void proj_wmma(const float* __restrict__ X,
                          const float* __restrict__ W,
                          float* __restrict__ Y,
                          int Kd, int N) {
    const int rowbase = blockIdx.x * 64;
    const int tn   = blockIdx.y;
    const int lane = threadIdx.x;
    const int half = lane >> 4;
    const int l16  = lane & 15;
    const int colB = tn * 16 + l16;

    v8f acc[4] = {};
    for (int k0 = 0; k0 < Kd; k0 += 32) {
        v16bf b;
        const float* bp = W + (size_t)(k0 + half * 8) * N + colB;
#pragma unroll
        for (int i = 0; i < 16; ++i)
            b[i] = (__bf16)bp[(size_t)((i & 7) + 16 * (i >> 3)) * N];

#pragma unroll
        for (int m = 0; m < 4; ++m) {
            const float* ap = X + (size_t)(rowbase + m * 16 + l16) * Kd + k0 + half * 8;
            v16bf a;
#pragma unroll
            for (int i = 0; i < 8; ++i) {
                a[i]     = (__bf16)ap[i];
                a[8 + i] = (__bf16)ap[16 + i];
            }
            acc[m] = __builtin_amdgcn_wmma_f32_16x16x32_bf16(
                         false, a, false, b, (short)0, acc[m], false, false);
        }
    }
#pragma unroll
    for (int m = 0; m < 4; ++m)
#pragma unroll
        for (int r = 0; r < 8; ++r)
            Y[(size_t)(rowbase + m * 16 + r + 8 * half) * N + tn * 16 + l16] = acc[m][r];
}

// ---------------------------------------------------------------------------
// Kernel 2: fused additive scores + masked softmax.
// Block = 256 threads = 16 queries x all 1024 keys of one batch.
// kf streamed via TDM (tensor_load_to_lds, double-buffered, TENSORcnt);
// q-rows / w_v staged via async-to-LDS (ASYNCcnt); 16x1024 f32 scores in LDS.
// ---------------------------------------------------------------------------
#define KT    64          // k rows per LDS tile
#define KPAD  132         // padded LDS row pitch (floats); 132 % 64 = 4
#define NT    (cLK / KT)  // 16 tiles

__global__ void scores_softmax(const float* __restrict__ qf,
                               const float* __restrict__ kf,
                               const float* __restrict__ wv,
                               const int*   __restrict__ vlen,
                               __bf16* __restrict__ attn) {
    extern __shared__ char smem[];
    float* skf  = (float*)smem;                       // 2 * KT * KPAD   (67.6 KB)
    float* ssc  = skf + 2 * KT * KPAD;                // 16 * 1024       (64 KB)
    float* sq   = ssc + 16 * cLK;                     // 16 * cH         (8 KB)
    float* sw   = sq + 16 * cH;                       // cH              (0.5 KB)
    float* sred = sw + cH;                            // 256             (1 KB)

    const int b  = blockIdx.x >> 4;
    const int q0 = (blockIdx.x & 15) * 16;
    const int t  = threadIdx.x;
    const int q  = t >> 4;                 // 0..15
    const int l  = t & 15;                 // 0..15

    const float* kfb = kf + (size_t)b * cLK * cH;
    const float* qfb = qf + ((size_t)b * cLQ + q0) * cH;

    // TDM prefetch of kf tile 0 (wave 0 only; EXEC uniform per wave)
    if (t < 32) tdm_load_tile(skf, kfb);

    // async-to-LDS staging of 16 q-rows (2 x b128 per thread) and w_v
#pragma unroll
    for (int j = 0; j < 2; ++j) {
        int f = t + 256 * j;               // float4 index, 512 total
        async_copy_b128(sq + f * 4, qfb + f * 4);
    }
    if (t < 32) async_copy_b128(sw + t * 4, wv + t * 4);
    S_WAIT_ASYNCCNT(0);                    // own q/w copies done

    const int VL = vlen[b];
    constexpr float LOG2E = 1.44269504088896340736f;

    for (int T = 0; T < NT; ++T) {
        const int buf = T & 1;
        if (t < 32) {                      // wave 0 drives the TDM pipeline
            if (T + 1 < NT) {
                tdm_load_tile(skf + (1 - buf) * KT * KPAD,
                              kfb + (size_t)(T + 1) * KT * cH);
                wait_tensorcnt<1>();       // tile T complete, T+1 in flight
            } else {
                wait_tensorcnt<0>();
            }
        }
        __syncthreads();                   // publish tile T to all waves

        const float* kt = skf + buf * KT * KPAD;
#pragma unroll
        for (int j = 0; j < 4; ++j) {
            const int klocal = l + 16 * j;
            const int k = T * KT + klocal;
            const float* kp = kt + klocal * KPAD;
            const float* qp = sq + q * cH;
            float acc = 0.f;
#pragma unroll 8
            for (int h = 0; h < cH; h += 4) {
                float4 kv = *(const float4*)(kp + h);
                float4 qv = *(const float4*)(qp + h);
                float4 wq = *(const float4*)(sw + h);
                acc += fast_tanh(qv.x + kv.x) * wq.x;
                acc += fast_tanh(qv.y + kv.y) * wq.y;
                acc += fast_tanh(qv.z + kv.z) * wq.z;
                acc += fast_tanh(qv.w + kv.w) * wq.w;
            }
            ssc[q * cLK + k] = (k < VL) ? acc : cNEG;
        }
        __syncthreads();                   // tile buffer free for T+2 DMA
    }

    // ---- fused masked softmax over each q row (16 lanes per row) ----
    float* row = ssc + q * cLK;
    float m = -3.0e38f;
#pragma unroll 16
    for (int i = 0; i < 64; ++i) m = fmaxf(m, row[l + 16 * i]);
    sred[t] = m;
    __syncthreads();
    for (int off = 8; off > 0; off >>= 1) {
        if (l < off) sred[t] = fmaxf(sred[t], sred[t + off]);
        __syncthreads();
    }
    m = sred[q * 16];
    __syncthreads();

    float lsum = 0.f;
#pragma unroll 16
    for (int i = 0; i < 64; ++i) {
        float e = fast_exp2((row[l + 16 * i] - m) * LOG2E);
        row[l + 16 * i] = e;
        lsum += e;
    }
    sred[t] = lsum;
    __syncthreads();
    for (int off = 8; off > 0; off >>= 1) {
        if (l < off) sred[t] += sred[t + off];
        __syncthreads();
    }
    const float inv = 1.0f / sred[q * 16];

    __bf16* arow = attn + ((size_t)b * cLQ + q0 + q) * cLK;
#pragma unroll 16
    for (int i = 0; i < 64; ++i)
        arow[l + 16 * i] = (__bf16)(row[l + 16 * i] * inv);
}

// ---------------------------------------------------------------------------
// Kernel 3: out[b] = attn[b] @ V[b] via bf16 WMMA, 64x16 strip per wave.
// ---------------------------------------------------------------------------
__global__ void av_wmma(const __bf16* __restrict__ attn,
                        const float*  __restrict__ V,
                        float* __restrict__ out) {
    const int b    = blockIdx.z;
    const int rowb = blockIdx.x * 64;
    const int tn   = blockIdx.y;
    const int lane = threadIdx.x;
    const int half = lane >> 4;
    const int l16  = lane & 15;

    const float* vb = V + (size_t)b * cLK * cD + tn * 16 + l16;
    const __bf16* ab = attn + (size_t)b * cLQ * cLK;

    v8f acc[4] = {};
    for (int k0 = 0; k0 < cLK; k0 += 32) {
        v16bf bm;
#pragma unroll
        for (int i = 0; i < 16; ++i)
            bm[i] = (__bf16)vb[(size_t)(k0 + half * 8 + (i & 7) + 16 * (i >> 3)) * cD];

#pragma unroll
        for (int m = 0; m < 4; ++m) {
            const __bf16* ap = ab + (size_t)(rowb + m * 16 + l16) * cLK + k0 + half * 8;
            v16bf a;
#pragma unroll
            for (int i = 0; i < 8; ++i) {
                a[i]     = ap[i];
                a[8 + i] = ap[16 + i];
            }
            acc[m] = __builtin_amdgcn_wmma_f32_16x16x32_bf16(
                         false, a, false, bm, (short)0, acc[m], false, false);
        }
    }
    float* ob = out + (size_t)b * cLQ * cD;
#pragma unroll
    for (int m = 0; m < 4; ++m)
#pragma unroll
        for (int r = 0; r < 8; ++r)
            ob[(size_t)(rowb + m * 16 + r + 8 * half) * cD + tn * 16 + l16] = acc[m][r];
}

// ---------------------------------------------------------------------------
extern "C" void kernel_launch(void* const* d_in, const int* in_sizes, int n_in,
                              void* d_out, int out_size, void* d_ws, size_t ws_size,
                              hipStream_t stream) {
    const float* Q    = (const float*)d_in[0];   // (B, LQ, DQ)
    const float* K    = (const float*)d_in[1];   // (B, LK, DK)
    const float* V    = (const float*)d_in[2];   // (B, LK, DV)
    const int*   vlen = (const int*)  d_in[3];   // (B,)
    const float* Wq   = (const float*)d_in[4];   // (DQ, H)
    const float* Wk   = (const float*)d_in[5];   // (DK, H)
    const float* wv   = (const float*)d_in[6];   // (H,)
    float* out = (float*)d_out;                  // (B, LQ, DV)

    // workspace: qf | kf | attn(bf16)   (~9 MB)
    float*  qf   = (float*)d_ws;                          // B*LQ*H
    float*  kf   = qf + (size_t)cB * cLQ * cH;            // B*LK*H
    __bf16* attn = (__bf16*)(kf + (size_t)cB * cLK * cH); // B*LQ*LK

    // 1) projections (WMMA bf16, f32 accum); 64-row strips
    proj_wmma<<<dim3(cB * cLQ / 64, cH / 16), dim3(32), 0, stream>>>(Q, Wq, qf, cD, cH);
    proj_wmma<<<dim3(cB * cLK / 64, cH / 16), dim3(32), 0, stream>>>(K, Wk, kf, cD, cH);

    // 2) fused tanh scores + masked softmax; TDM kf streaming + async q/w staging
    size_t smem = (2 * KT * KPAD + 16 * cLK + 16 * cH + cH + 256) * sizeof(float);
    scores_softmax<<<dim3(cB * (cLQ / 16)), dim3(256), smem, stream>>>(qf, kf, wv, vlen, attn);

    // 3) attn @ V (WMMA bf16, f32 accum); 64-row strips
    av_wmma<<<dim3(cLQ / 64, cD / 16, cB), dim3(32), 0, stream>>>(attn, V, out);
}
